// Indexer_1563368095775
// MI455X (gfx1250) — matrix operand order repaired and checked by reading, not compiled, not run on hardware
//
#include <hip/hip_runtime.h>
#include <hip/hip_bf16.h>
#include <stdint.h>

#define T_SEQ 2048
#define DIN   2048
#define DKV   512
#define NH    32
#define HD    64
#define RD    32
#define NOPE  32
#define TOPK  512
#define LN_EPS 1e-5f
// ln(10000)/16
#define ROPE_LOGC 0.5756462732485114f

typedef __bf16 bf16_t;
typedef bf16_t v16bf __attribute__((ext_vector_type(16)));
typedef float  v8f   __attribute__((ext_vector_type(8)));
typedef uint32_t u32;
typedef u32 v4u __attribute__((ext_vector_type(4)));
typedef u32 v8u __attribute__((ext_vector_type(8)));
typedef unsigned long long u64;

__device__ __forceinline__ unsigned short f2bf(float f) {
    u32 u = __builtin_bit_cast(u32, f);
    u32 r = u + 0x7FFFu + ((u >> 16) & 1u);   // round-to-nearest-even
    return (unsigned short)(r >> 16);
}

// B fragment: 16 contiguous bf16 (32B) per lane
__device__ __forceinline__ v16bf load32B(const unsigned short* p) {
    v8u u = *(const v8u*)p;
    return __builtin_bit_cast(v16bf, u);
}
// A fragment: two 16B chunks per lane (K {0..15} or {16..31})
__device__ __forceinline__ v16bf load2x16B(const unsigned short* plo,
                                           const unsigned short* phi) {
    v4u lo = *(const v4u*)plo;
    v4u hi = *(const v4u*)phi;
    v8u u;
    u[0]=lo[0]; u[1]=lo[1]; u[2]=lo[2]; u[3]=lo[3];
    u[4]=hi[0]; u[5]=hi[1]; u[6]=hi[2]; u[7]=hi[3];
    return __builtin_bit_cast(v16bf, u);
}

// ---------------------------------------------------------------- converts
__global__ void cvt_x_kernel(const float* __restrict__ x,
                             unsigned short* __restrict__ xbf, int n) {
    int i = blockIdx.x * blockDim.x + threadIdx.x;
    if (i < n) xbf[i] = f2bf(x[i]);
}

// Wq (K x N, row-major) -> WqT bf16 (N x K) via LDS tile transpose
__global__ void cvt_wqT_kernel(const float* __restrict__ Wq,
                               unsigned short* __restrict__ wqT) {
    __shared__ float tile[32][33];
    int tx = threadIdx.x, ty = threadIdx.y;
    int k0 = blockIdx.x * 32, n0 = blockIdx.y * 32;
    for (int r = ty; r < 32; r += 8)
        tile[r][tx] = Wq[(k0 + r) * DIN + n0 + tx];
    __syncthreads();
    for (int r = ty; r < 32; r += 8)
        wqT[(n0 + r) * DIN + (k0 + tx)] = f2bf(tile[tx][r]);
}

// ---------------------------------------------------------------- k path
// k = RoPE(LN(kv @ Wk + b)) ; 8 rows per block, 512 threads (row=tid/64,col=tid&63)
__global__ void k_proj_kernel(const float* __restrict__ kv,
                              const float* __restrict__ Wk,
                              const float* __restrict__ Wkb,
                              const float* __restrict__ kg,
                              const float* __restrict__ kb,
                              float* __restrict__ kf_out,
                              unsigned short* __restrict__ kfb) {
    __shared__ float skv[8][DKV];
    __shared__ float red[8][HD];
    __shared__ float sval[8][HD];
    int tid = threadIdx.x;
    int t0 = blockIdx.x * 8;
    for (int it = 0; it < 8; ++it) {
        int lin = tid + it * 512;
        int rr = lin >> 9, cc = lin & 511;
        skv[rr][cc] = kv[(t0 + rr) * DKV + cc];
    }
    __syncthreads();
    int r = tid >> 6, c = tid & 63;
    float acc = Wkb[c];
    for (int i = 0; i < DKV; ++i) acc += skv[r][i] * Wk[i * HD + c];
    red[r][c] = acc; __syncthreads();
    for (int s = 32; s > 0; s >>= 1) { if (c < s) red[r][c] += red[r][c + s]; __syncthreads(); }
    float mu = red[r][0] * (1.f / 64.f);
    __syncthreads();
    float d = acc - mu;
    red[r][c] = d * d; __syncthreads();
    for (int s = 32; s > 0; s >>= 1) { if (c < s) red[r][c] += red[r][c + s]; __syncthreads(); }
    float var = red[r][0] * (1.f / 64.f);
    float val = d * rsqrtf(var + LN_EPS) * kg[c] + kb[c];
    sval[r][c] = val; __syncthreads();
    float out = val;
    if (c >= NOPE) {
        int i2 = c - NOPE, i = i2 >> 1;
        float ang = (float)(t0 + r) * __expf(-(float)i * ROPE_LOGC);
        float cs = __cosf(ang), sn = __sinf(ang);
        float v1 = sval[r][NOPE + 2 * i], v2 = sval[r][NOPE + 2 * i + 1];
        out = (i2 & 1) ? (v1 * sn + v2 * cs) : (v1 * cs - v2 * sn);
    }
    int t = t0 + r;
    kf_out[t * HD + c] = out;
    kfb[t * HD + c]    = f2bf(out);
}

// ---------------------------------------------------------------- w path
// w = LN(x @ Ww + b); one wave32 per row, lane = head
__global__ void w_proj_kernel(const float* __restrict__ x,
                              const float* __restrict__ Ww,
                              const float* __restrict__ Wwb,
                              const float* __restrict__ wg,
                              const float* __restrict__ wb,
                              float* __restrict__ w_out) {
    int tid = threadIdx.x;
    int t = blockIdx.x * 8 + (tid >> 5);
    int h = tid & 31;
    const float* xr = x + t * DIN;
    float acc = Wwb[h];
    for (int k = 0; k < DIN; ++k) acc += xr[k] * Ww[k * NH + h];
    float s = acc;
    for (int m = 16; m > 0; m >>= 1) s += __shfl_xor(s, m, 32);
    float mu = s * (1.f / 32.f);
    float d = acc - mu;
    float v = d * d;
    for (int m = 16; m > 0; m >>= 1) v += __shfl_xor(v, m, 32);
    float var = v * (1.f / 32.f);
    w_out[t * NH + h] = d * rsqrtf(var + LN_EPS) * wg[h] + wb[h];
}

// ---------------------------------------------------------------- q GEMM (bf16 WMMA)
// 16x16 tile per wave, K=2048 in steps of 32 -> 64 v_wmma per wave
__global__ void q_gemm_kernel(const unsigned short* __restrict__ xbf,
                              const unsigned short* __restrict__ wqT,
                              const float* __restrict__ Wqb,
                              float* __restrict__ q32) {
    int wid = threadIdx.x >> 5;
    int lane = threadIdx.x & 31;
    int gwave = blockIdx.x * 8 + wid;
    int tile_t = gwave >> 7;
    int tile_n = gwave & 127;
    int m = lane & 15, hi = lane >> 4;
    const unsigned short* arow = xbf + (tile_t * 16 + m) * DIN;
    const unsigned short* brow = wqT + (tile_n * 16 + m) * DIN;
    v8f acc = {};
    for (int kk = 0; kk < DIN; kk += 32) {
        v16bf a = load2x16B(arow + kk + 16 * hi, arow + kk + 8 + 16 * hi);
        v16bf b = load32B(brow + kk + 16 * hi);
        acc = __builtin_amdgcn_wmma_f32_16x16x32_bf16(false, a, false, b,
                                                      (short)0, acc, false, false);
        if (kk + 64 < DIN) {
            __builtin_prefetch(arow + kk + 64, 0, 1);
            __builtin_prefetch(brow + kk + 64, 0, 1);
        }
    }
    int col = tile_n * 16 + (lane & 15);
    float bias = Wqb[col];
#pragma unroll
    for (int r = 0; r < 8; ++r) {
        int row = tile_t * 16 + r + 8 * hi;
        q32[row * DIN + col] = acc[r] + bias;
    }
}

// ---------------------------------------------------------------- q LN + RoPE -> bf16 [t][h*64+d]
__global__ void q_ln_rope_kernel(const float* __restrict__ q32,
                                 const float* __restrict__ qg,
                                 const float* __restrict__ qb,
                                 unsigned short* __restrict__ qfb) {
    __shared__ float ssum[256], ssq[256];
    int t = blockIdx.x, tid = threadIdx.x;
    const float* row = q32 + t * DIN;
    float s = 0.f, sq = 0.f;
    for (int i = tid; i < DIN; i += 256) { float v = row[i]; s += v; sq += v * v; }
    ssum[tid] = s; ssq[tid] = sq; __syncthreads();
    for (int st = 128; st > 0; st >>= 1) {
        if (tid < st) { ssum[tid] += ssum[tid + st]; ssq[tid] += ssq[tid + st]; }
        __syncthreads();
    }
    float mu = ssum[0] * (1.f / (float)DIN);
    float var = ssq[0] * (1.f / (float)DIN) - mu * mu;
    float rstd = rsqrtf(var + LN_EPS);
    for (int j = tid; j < DIN; j += 256) {
        int h = j >> 6, d = j & 63;
        float out;
        if (d < NOPE) {
            int p = h * NOPE + d;
            out = (row[p] - mu) * rstd * qg[p] + qb[p];
        } else {
            int i2 = d - NOPE, i = i2 >> 1;
            int p0 = NH * NOPE + h * RD + 2 * i;
            int p1 = p0 + 1;
            float v1 = (row[p0] - mu) * rstd * qg[p0] + qb[p0];
            float v2 = (row[p1] - mu) * rstd * qg[p1] + qb[p1];
            float ang = (float)t * __expf(-(float)i * ROPE_LOGC);
            float cs = __cosf(ang), sn = __sinf(ang);
            out = (i2 & 1) ? (v1 * sn + v2 * cs) : (v1 * cs - v2 * sn);
        }
        qfb[t * DIN + h * HD + d] = f2bf(out);
    }
}

// ---------------------------------------------------------------- fused scores
// idx_scores[t,k] = sum_h w[t,h]*relu(q_full[t,h,:].k_full[k,:]) with causal mask.
// One wave per 16x16 tile; 32 heads x 2 WMMA(16x16x32 bf16); relu+weight in regs.
__global__ void scores_kernel(const unsigned short* __restrict__ qfb,
                              const unsigned short* __restrict__ kfb,
                              const float* __restrict__ wbuf,
                              float* __restrict__ out) {
    __shared__ float sw[16 * NH];
    int tid = threadIdx.x;
    int wid = tid >> 5, lane = tid & 31;
    int tile_t = blockIdx.y;
    int tile_k = blockIdx.x * 8 + wid;
    for (int i = tid; i < 16 * NH; i += 256)
        sw[i] = wbuf[(tile_t * 16 + (i >> 5)) * NH + (i & 31)];
    __syncthreads();
    int m = lane & 15, hi = lane >> 4;
    const unsigned short* qrow = qfb + (tile_t * 16 + m) * DIN;
    const unsigned short* krow = kfb + (tile_k * 16 + m) * HD;
    v16bf b0 = load32B(krow + 16 * hi);        // d = 0..31 chunk
    v16bf b1 = load32B(krow + 32 + 16 * hi);   // d = 32..63 chunk
    v8f acc = {};
    for (int h = 0; h < NH; ++h) {
        const unsigned short* qh = qrow + h * HD;
        v16bf a0 = load2x16B(qh + 16 * hi, qh + 8 + 16 * hi);
        v16bf a1 = load2x16B(qh + 32 + 16 * hi, qh + 40 + 16 * hi);
        v8f ch = {};
        ch = __builtin_amdgcn_wmma_f32_16x16x32_bf16(false, a0, false, b0,
                                                     (short)0, ch, false, false);
        ch = __builtin_amdgcn_wmma_f32_16x16x32_bf16(false, a1, false, b1,
                                                     (short)0, ch, false, false);
#pragma unroll
        for (int r = 0; r < 8; ++r) {
            float wv = sw[(r + 8 * hi) * NH + h];
            float v = ch[r];
            acc[r] += (v > 0.f ? v : 0.f) * wv;
        }
    }
    int col = tile_k * 16 + (lane & 15);
#pragma unroll
    for (int r = 0; r < 8; ++r) {
        int row = tile_t * 16 + r + 8 * hi;
        out[row * T_SEQ + col] = (col <= row) ? acc[r] : -1.0e9f;
    }
}

// ---------------------------------------------------------------- top-k (bitonic in LDS)
// key = (~orderable(score) << 32) | index : ascending sort => desc score, ties -> low idx
__global__ void topk_kernel(const float* __restrict__ scores,
                            float* __restrict__ out_idx) {
    __shared__ u64 sk[T_SEQ];
    int t = blockIdx.x, tid = threadIdx.x;
    const float* srow = scores + t * T_SEQ;
    for (int k = tid; k < T_SEQ; k += 256) {
        u32 f = __builtin_bit_cast(u32, srow[k]);
        u32 ord = f ^ ((f >> 31) ? 0xFFFFFFFFu : 0x80000000u);
        sk[k] = ((u64)(~ord) << 32) | (u32)k;
    }
    __syncthreads();
    for (int ksz = 2; ksz <= T_SEQ; ksz <<= 1) {
        for (int j = ksz >> 1; j > 0; j >>= 1) {
            for (int e = tid; e < T_SEQ; e += 256) {
                int ixj = e ^ j;
                if (ixj > e) {
                    bool up = ((e & ksz) == 0);
                    u64 a = sk[e], b = sk[ixj];
                    if ((a > b) == up) { sk[e] = b; sk[ixj] = a; }
                }
            }
            __syncthreads();
        }
    }
    for (int jj = tid; jj < TOPK; jj += 256)
        out_idx[t * TOPK + jj] = (float)(u32)(sk[jj] & 0xFFFFFFFFull);
}

// ----------------------------------------------------------------
extern "C" void kernel_launch(void* const* d_in, const int* in_sizes, int n_in,
                              void* d_out, int out_size, void* d_ws, size_t ws_size,
                              hipStream_t stream) {
    const float* x    = (const float*)d_in[0];
    const float* kv   = (const float*)d_in[1];
    /* d_in[2] = mask (causal tril, computed analytically) */
    const float* Wq_w = (const float*)d_in[3];
    const float* Wq_b = (const float*)d_in[4];
    const float* Wk_w = (const float*)d_in[5];
    const float* Wk_b = (const float*)d_in[6];
    const float* Ww_w = (const float*)d_in[7];
    const float* Ww_b = (const float*)d_in[8];
    const float* q_g  = (const float*)d_in[9];
    const float* q_b  = (const float*)d_in[10];
    const float* k_g  = (const float*)d_in[11];
    const float* k_b  = (const float*)d_in[12];
    const float* w_g  = (const float*)d_in[13];
    const float* w_b  = (const float*)d_in[14];

    float* out        = (float*)d_out;
    float* out_idx    = out;                          // 2048*512
    float* out_scores = out + T_SEQ * TOPK;           // 2048*2048
    float* out_kfull  = out_scores + T_SEQ * T_SEQ;   // 2048*64

    char* ws = (char*)d_ws;
    unsigned short* xbf = (unsigned short*)(ws);                          // 8 MB
    unsigned short* wqT = (unsigned short*)(ws + (size_t)8  * 1048576);   // 8 MB
    float*          q32 = (float*)         (ws + (size_t)16 * 1048576);   // 16 MB
    unsigned short* qfb = (unsigned short*)(ws + (size_t)32 * 1048576);   // 8 MB
    unsigned short* kfb = (unsigned short*)(ws + (size_t)40 * 1048576);   // 256 KB
    float*          wws = (float*)         (ws + (size_t)40 * 1048576 + 524288); // 256 KB

    cvt_x_kernel   <<<(T_SEQ * DIN) / 256, 256, 0, stream>>>(x, xbf, T_SEQ * DIN);
    cvt_wqT_kernel <<<dim3(64, 64), dim3(32, 8), 0, stream>>>(Wq_w, wqT);
    k_proj_kernel  <<<T_SEQ / 8, 512, 0, stream>>>(kv, Wk_w, Wk_b, k_g, k_b,
                                                   out_kfull, kfb);
    w_proj_kernel  <<<T_SEQ / 8, 256, 0, stream>>>(x, Ww_w, Ww_b, w_g, w_b, wws);
    q_gemm_kernel  <<<(128 * 128) / 8, 256, 0, stream>>>(xbf, wqT, Wq_b, q32);
    q_ln_rope_kernel<<<T_SEQ, 256, 0, stream>>>(q32, q_g, q_b, qfb);
    scores_kernel  <<<dim3(16, 128), 256, 0, stream>>>(qfb, kfb, wws, out_scores);
    topk_kernel    <<<T_SEQ, 256, 0, stream>>>(out_scores, out_idx);
}
